// TemporalVirtualNodePredictor_18021682774307
// MI455X (gfx1250) — matrix operand order repaired and checked by reading, not compiled, not run on hardware
//
#include <hip/hip_runtime.h>
#include <hip/hip_bf16.h>

// ---------------------------------------------------------------------------
// Problem constants (from the reference)
// ---------------------------------------------------------------------------
#define NNODES 3072
#define EMB    512
#define HEADS  8
#define HDIM   64
#define NEDGE  49152
#define NN     ((long long)NNODES * (long long)NNODES)

typedef __bf16 bf16;
typedef __bf16 v16bf __attribute__((ext_vector_type(16)));
typedef float  v8f   __attribute__((ext_vector_type(8)));

__device__ __forceinline__ bf16 f2bf(float f) { return (bf16)f; }

// ---------------------------------------------------------------------------
// Tiled WMMA GEMM:  C = alpha * A(bf16) @ B(bf16) [+ bias] ; optional +=
// B is ALWAYS supplied "column-major": element (k,n) lives at Bg[n*ldb + k].
// blockIdx.z selects a "head slice" via element strides sAh/sBh/sCh/sCbh.
// Tile: 128x64 per 256-thread block; 8 waves, each wave owns a 32x32 quad of
// 16x16 WMMA tiles; BK = 32 (one v_wmma_f32_16x16x32_bf16 per frag pair).
// Register double-buffering: tile k+1 is loaded from global into VGPRs while
// the WMMAs consume tile k from LDS; +prefetch of tile k+2.
// LDS layouts keep each lane's fragment data contiguous (-> ds_load_b128):
//   As[m][k] : A-frag lane(m) reads k-chunks [8*lg,+8) and [16+8*lg,+8)
//   Bs[n][k] : B-frag lane(n) reads k-chunks [8*lg,+8) and [16+8*lg,+8)
// ---------------------------------------------------------------------------
#define BM 128
#define BN 64
#define BK 32
#define LDA_S (BK + 8)   // elems; row offset = 80B -> 16B aligned
#define LDB_S (BK + 8)

__global__ __launch_bounds__(256)
void wmma_gemm(const bf16* __restrict__ Ag, int lda, long long sAh,
               const bf16* __restrict__ Bg, int ldb, long long sBh,
               float* Cf, int ldc, long long sCh,
               bf16*  Cb, long long ldcb, long long sCbh, int cbT,
               const float* __restrict__ bias,
               int Kd, float alpha, int accumulate)
{
    __shared__ bf16 As[BM * LDA_S];
    __shared__ bf16 Bs[BN * LDB_S];

    const int tid  = threadIdx.x;
    const int lane = tid & 31;
    const int wid  = tid >> 5;
    const int wm   = (wid & 3) * 32;   // wave row offset inside block tile
    const int wn   = (wid >> 2) * 32;  // wave col offset inside block tile
    const int l15  = lane & 15;
    const int lg   = lane >> 4;        // lane group (0: lanes 0-15, 1: 16-31)

    const long long tileM = (long long)blockIdx.y * BM;
    const long long tileN = (long long)blockIdx.x * BN;

    const bf16* A = Ag + (long long)blockIdx.z * sAh;
    const bf16* B = Bg + (long long)blockIdx.z * sBh;

    // per-thread staging assignment: 2 A-chunks + 1 B-chunk of 8 bf16 each
    const int ar = tid >> 2;           // A rows ar and ar+64
    const int ac = (tid & 3) * 8;      // A column-chunk
    const int bn = tid >> 2;           // B row (n)
    const int bk = (tid & 3) * 8;      // B k-chunk

    const bf16* aptr0 = A + (tileM + ar) * lda + ac;
    const bf16* aptr1 = A + (tileM + ar + 64) * lda + ac;
    const bf16* bptr  = B + (tileN + bn) * ldb + bk;

    uint4 aR0 = *(const uint4*)aptr0;
    uint4 aR1 = *(const uint4*)aptr1;
    uint4 bR  = *(const uint4*)bptr;

    v8f acc[2][2] = {};

    for (int k0 = 0; k0 < Kd; k0 += BK) {
        // ---- dump staged registers into LDS ----
        *(uint4*)&As[ar * LDA_S + ac]        = aR0;
        *(uint4*)&As[(ar + 64) * LDA_S + ac] = aR1;
        *(uint4*)&Bs[bn * LDB_S + bk]        = bR;
        __syncthreads();

        // ---- kick off next tile's global loads (overlap with WMMA) ----
        if (k0 + BK < Kd) {
            aR0 = *(const uint4*)(aptr0 + k0 + BK);
            aR1 = *(const uint4*)(aptr1 + k0 + BK);
            bR  = *(const uint4*)(bptr  + k0 + BK);
            if (k0 + 2 * BK < Kd) {
                __builtin_prefetch(aptr0 + k0 + 2 * BK, 0, 1);
                __builtin_prefetch(aptr1 + k0 + 2 * BK, 0, 1);
                __builtin_prefetch(bptr  + k0 + 2 * BK, 0, 1);
            }
        }

        // ---- build fragments (contiguous per-lane chunks in LDS) ----
        v16bf afr[2], bfr[2];
        #pragma unroll
        for (int f = 0; f < 2; ++f) {
            const bf16* ap = &As[(wm + f * 16 + l15) * LDA_S];
            #pragma unroll
            for (int t = 0; t < 8; ++t) {
                afr[f][t]     = ap[8 * lg + t];
                afr[f][8 + t] = ap[16 + 8 * lg + t];
            }
            const bf16* bp = &Bs[(wn + f * 16 + l15) * LDB_S];
            #pragma unroll
            for (int t = 0; t < 8; ++t) {
                bfr[f][t]     = bp[8 * lg + t];
                bfr[f][8 + t] = bp[16 + 8 * lg + t];
            }
        }
        #pragma unroll
        for (int i = 0; i < 2; ++i)
            #pragma unroll
            for (int j = 0; j < 2; ++j)
                acc[i][j] = __builtin_amdgcn_wmma_f32_16x16x32_bf16(
                    false, afr[i], false, bfr[j], (short)0, acc[i][j],
                    false, false);
        __syncthreads();
    }

    // ---- epilogue: C/D layout = VGPR r -> row r + 8*lanegroup, col lane&15 ----
    const long long zz = (long long)blockIdx.z;
    #pragma unroll
    for (int i = 0; i < 2; ++i) {
        #pragma unroll
        for (int j = 0; j < 2; ++j) {
            const long long row0 = tileM + wm + i * 16 + 8 * lg;
            const long long col  = tileN + wn + j * 16 + l15;
            const float bcol = bias ? bias[col] : 0.0f;
            if (Cf) {
                float* p = Cf + zz * sCh + row0 * ldc + col;
                #pragma unroll
                for (int r = 0; r < 8; ++r) {
                    float v = alpha * acc[i][j][r] + bcol;
                    if (accumulate) p[(long long)r * ldc] += v;
                    else            p[(long long)r * ldc]  = v;
                }
            }
            if (Cb) {
                if (cbT) {   // transposed: 8 contiguous bf16 per lane
                    bf16* p = Cb + zz * sCbh + col * ldcb + row0;
                    #pragma unroll
                    for (int r = 0; r < 8; ++r)
                        p[r] = f2bf(alpha * acc[i][j][r] + bcol);
                } else {
                    bf16* p = Cb + zz * sCbh + row0 * ldcb + col;
                    #pragma unroll
                    for (int r = 0; r < 8; ++r)
                        p[(long long)r * ldcb] = f2bf(alpha * acc[i][j][r] + bcol);
                }
            }
        }
    }
}

// ---------------------------------------------------------------------------
// Elementwise f32 -> bf16 conversion
// ---------------------------------------------------------------------------
__global__ void cvt_f32_bf16(const float* __restrict__ s, bf16* __restrict__ d,
                             long long n)
{
    long long i = (long long)blockIdx.x * 256 + threadIdx.x;
    if (i < n) d[i] = f2bf(s[i]);
}

// Transpose-convert a [EMB,EMB] f32 matrix into bf16 [n][k] (column-major B).
__global__ void cvt_transpose_bf16(const float* __restrict__ s,
                                   bf16* __restrict__ d)
{
    __shared__ float t[32][33];
    int bx = blockIdx.x * 32, by = blockIdx.y * 32;
    int x = threadIdx.x, y0 = threadIdx.y;           // block (32,8)
    #pragma unroll
    for (int dy = 0; dy < 32; dy += 8)
        t[y0 + dy][x] = s[(long long)(by + y0 + dy) * EMB + bx + x];
    __syncthreads();
    #pragma unroll
    for (int dy = 0; dy < 32; dy += 8)
        d[(long long)(bx + y0 + dy) * EMB + by + x] = f2bf(t[x][y0 + dy]);
}

// Gather temporal embeddings by time_index and convert to bf16
__global__ void gather_te(const float* __restrict__ te,
                          const int* __restrict__ tidx,
                          bf16* __restrict__ d)
{
    int i = blockIdx.x;
    long long src = (long long)tidx[i] * EMB;
    long long dst = (long long)i * EMB;
    for (int c = threadIdx.x; c < EMB; c += 256)
        d[dst + c] = f2bf(te[src + c]);
}

// Per-edge bias: (edge_attr @ We + be) scattered into dense scores via atomics
__global__ void edge_scatter(const float* __restrict__ edge_attr,
                             const float* __restrict__ We,   // [4,HEADS]
                             const float* __restrict__ be,   // [HEADS]
                             const int*   __restrict__ eidx, // [2,E] flat
                             float* __restrict__ scores)     // [H,N,N]
{
    int e = blockIdx.x * 256 + threadIdx.x;
    if (e >= NEDGE) return;
    float a0 = edge_attr[e * 4 + 0], a1 = edge_attr[e * 4 + 1];
    float a2 = edge_attr[e * 4 + 2], a3 = edge_attr[e * 4 + 3];
    int src = eidx[e];
    int dst = eidx[NEDGE + e];
    long long base = (long long)src * NNODES + dst;
    #pragma unroll
    for (int h = 0; h < HEADS; ++h) {
        float b = be[h] + a0 * We[0 * HEADS + h] + a1 * We[1 * HEADS + h]
                        + a2 * We[2 * HEADS + h] + a3 * We[3 * HEADS + h];
        atomicAdd(scores + (long long)h * NN + base, b);
    }
}

// Masked softmax over one (head,row): scores f32 [H,N,N] -> attn bf16 [H,N,N]
__global__ __launch_bounds__(256)
void masked_softmax(const float* __restrict__ scores,
                    bf16* __restrict__ attn,
                    const int* __restrict__ batch)
{
    long long hi = blockIdx.x;               // h*N + i
    int i  = (int)(hi % NNODES);
    int bi = batch[i];
    const float* row = scores + hi * NNODES;
    bf16* out        = attn   + hi * NNODES;

    __shared__ float red[256];
    const int tid = threadIdx.x;
    float sv[NNODES / 256];                  // 12 cached values per thread

    float mloc = -__builtin_inff();
    #pragma unroll
    for (int t = 0; t < NNODES / 256; ++t) {
        int j = tid + t * 256;
        float s = row[j];
        if (batch[j] != bi) s = -__builtin_inff();
        sv[t] = s;
        mloc = fmaxf(mloc, s);
    }
    red[tid] = mloc; __syncthreads();
    for (int s = 128; s > 0; s >>= 1) {
        if (tid < s) red[tid] = fmaxf(red[tid], red[tid + s]);
        __syncthreads();
    }
    float m = red[0]; __syncthreads();

    float sloc = 0.f;
    #pragma unroll
    for (int t = 0; t < NNODES / 256; ++t) {
        float p = __expf(sv[t] - m);         // exp(-inf) = 0 handles mask
        sv[t] = p;
        sloc += p;
    }
    red[tid] = sloc; __syncthreads();
    for (int s = 128; s > 0; s >>= 1) {
        if (tid < s) red[tid] += red[tid + s];
        __syncthreads();
    }
    float inv = 1.0f / red[0];
    #pragma unroll
    for (int t = 0; t < NNODES / 256; ++t)
        out[tid + t * 256] = f2bf(sv[t] * inv);
}

// ---------------------------------------------------------------------------
// Launcher
// ---------------------------------------------------------------------------
extern "C" void kernel_launch(void* const* d_in, const int* in_sizes, int n_in,
                              void* d_out, int out_size, void* d_ws, size_t ws_size,
                              hipStream_t stream)
{
    (void)in_sizes; (void)n_in; (void)out_size; (void)ws_size;

    const float* x    = (const float*)d_in[0];
    const float* ea   = (const float*)d_in[1];
    const float* te   = (const float*)d_in[2];
    const float* Wq   = (const float*)d_in[3];  const float* bq  = (const float*)d_in[4];
    const float* Wk   = (const float*)d_in[5];  const float* bk  = (const float*)d_in[6];
    const float* Wv   = (const float*)d_in[7];  const float* bv  = (const float*)d_in[8];
    const float* Wo   = (const float*)d_in[9];  const float* bo  = (const float*)d_in[10];
    const float* We   = (const float*)d_in[11]; const float* be  = (const float*)d_in[12];
    const float* Wtq  = (const float*)d_in[13]; const float* btq = (const float*)d_in[14];
    const float* Wtk  = (const float*)d_in[15]; const float* btk = (const float*)d_in[16];
    const float* Wtv  = (const float*)d_in[17]; const float* btv = (const float*)d_in[18];
    const int*   eidx = (const int*)d_in[19];
    const int*   batch= (const int*)d_in[20];
    const int*   tidx = (const int*)d_in[21];
    float* out = (float*)d_out;

    // ---- carve workspace ----
    char* ws = (char*)d_ws;
    size_t off = 0;
    auto carve = [&](size_t bytes) {
        void* p = ws + off;
        off = (off + bytes + 255) & ~(size_t)255;
        return p;
    };
    float* scores = (float*)carve((size_t)HEADS * NN * sizeof(float));  // 302 MB
    bf16*  attn   = (bf16*) carve((size_t)HEADS * NN * sizeof(bf16));   // 151 MB
    bf16*  xb     = (bf16*) carve((size_t)NNODES * EMB * 2);
    bf16*  teb    = (bf16*) carve((size_t)NNODES * EMB * 2);
    bf16*  Qb     = (bf16*) carve((size_t)NNODES * EMB * 2);  // [i][h*64+d]
    bf16*  Kb     = (bf16*) carve((size_t)NNODES * EMB * 2);  // [j][h*64+d]
    bf16*  Vt     = (bf16*) carve((size_t)NNODES * EMB * 2);  // [h*64+d][j]
    bf16*  TQb    = (bf16*) carve((size_t)NNODES * EMB * 2);
    bf16*  TKb    = (bf16*) carve((size_t)NNODES * EMB * 2);
    bf16*  TVt    = (bf16*) carve((size_t)NNODES * EMB * 2);
    bf16*  Wqt    = (bf16*) carve((size_t)EMB * EMB * 2);     // all weights
    bf16*  Wkt    = (bf16*) carve((size_t)EMB * EMB * 2);     // transposed:
    bf16*  Wvt    = (bf16*) carve((size_t)EMB * EMB * 2);     // [n][k]
    bf16*  Wot    = (bf16*) carve((size_t)EMB * EMB * 2);
    bf16*  Wtqt   = (bf16*) carve((size_t)EMB * EMB * 2);
    bf16*  Wtkt   = (bf16*) carve((size_t)EMB * EMB * 2);
    bf16*  Wtvt   = (bf16*) carve((size_t)EMB * EMB * 2);
    float* comb   = (float*)carve((size_t)NNODES * EMB * sizeof(float));
    bf16*  combb  = (bf16*) carve((size_t)NNODES * EMB * 2);

    const long long nX = (long long)NNODES * EMB;
    const int cvtX = (int)((nX + 255) / 256);
    const dim3 gT(EMB / 32, EMB / 32, 1), bT(32, 8, 1);

    // ---- stage bf16 operands (weights transposed into B layout) ----
    cvt_transpose_bf16<<<gT, bT, 0, stream>>>(Wq,  Wqt);
    cvt_transpose_bf16<<<gT, bT, 0, stream>>>(Wk,  Wkt);
    cvt_transpose_bf16<<<gT, bT, 0, stream>>>(Wv,  Wvt);
    cvt_transpose_bf16<<<gT, bT, 0, stream>>>(Wo,  Wot);
    cvt_transpose_bf16<<<gT, bT, 0, stream>>>(Wtq, Wtqt);
    cvt_transpose_bf16<<<gT, bT, 0, stream>>>(Wtk, Wtkt);
    cvt_transpose_bf16<<<gT, bT, 0, stream>>>(Wtv, Wtvt);
    cvt_f32_bf16<<<cvtX, 256, 0, stream>>>(x, xb, nX);
    gather_te<<<NNODES, 256, 0, stream>>>(te, tidx, teb);

    const dim3 gLin(EMB / BN, NNODES / BM, 1);        // (8, 24, 1)
    const dim3 gScr(NNODES / BN, NNODES / BM, HEADS); // (48, 24, 8)
    const dim3 gAV(HDIM / BN, NNODES / BM, HEADS);    // (1, 24, 8)
    const float scale = 0.125f;                       // 1/sqrt(64)
    const long long NL = NNODES;

    // ---- QKV / temporal linears (bf16 out, fused bias; V/TV transposed) ----
    wmma_gemm<<<gLin, 256, 0, stream>>>(xb, EMB, 0, Wqt, EMB, 0,
        (float*)nullptr, 0, 0, Qb,  EMB, 0, 0, bq,  EMB, 1.0f, 0);
    wmma_gemm<<<gLin, 256, 0, stream>>>(xb, EMB, 0, Wkt, EMB, 0,
        (float*)nullptr, 0, 0, Kb,  EMB, 0, 0, bk,  EMB, 1.0f, 0);
    wmma_gemm<<<gLin, 256, 0, stream>>>(xb, EMB, 0, Wvt, EMB, 0,
        (float*)nullptr, 0, 0, Vt,  NL,  0, 1, bv,  EMB, 1.0f, 0);
    wmma_gemm<<<gLin, 256, 0, stream>>>(teb, EMB, 0, Wtqt, EMB, 0,
        (float*)nullptr, 0, 0, TQb, EMB, 0, 0, btq, EMB, 1.0f, 0);
    wmma_gemm<<<gLin, 256, 0, stream>>>(teb, EMB, 0, Wtkt, EMB, 0,
        (float*)nullptr, 0, 0, TKb, EMB, 0, 0, btk, EMB, 1.0f, 0);
    wmma_gemm<<<gLin, 256, 0, stream>>>(teb, EMB, 0, Wtvt, EMB, 0,
        (float*)nullptr, 0, 0, TVt, NL,  0, 1, btv, EMB, 1.0f, 0);

    // ---- spatial attention: scores = scale * Q @ K^T (per head) ----
    wmma_gemm<<<gScr, 256, 0, stream>>>(Qb, EMB, HDIM, Kb, EMB, HDIM,
        scores, NNODES, NN, (bf16*)nullptr, 0, 0, 0, (const float*)nullptr,
        HDIM, scale, 0);
    edge_scatter<<<(NEDGE + 255) / 256, 256, 0, stream>>>(ea, We, be, eidx, scores);
    masked_softmax<<<HEADS * NNODES, 256, 0, stream>>>(scores, attn, batch);
    wmma_gemm<<<gAV, 256, 0, stream>>>(attn, NNODES, NN, Vt, NNODES, (long long)HDIM * NNODES,
        comb, EMB, HDIM, (bf16*)nullptr, 0, 0, 0, (const float*)nullptr,
        NNODES, 1.0f, 0);

    // ---- temporal attention (reuse score/attn buffers) ----
    wmma_gemm<<<gScr, 256, 0, stream>>>(TQb, EMB, HDIM, TKb, EMB, HDIM,
        scores, NNODES, NN, (bf16*)nullptr, 0, 0, 0, (const float*)nullptr,
        HDIM, scale, 0);
    masked_softmax<<<HEADS * NNODES, 256, 0, stream>>>(scores, attn, batch);
    wmma_gemm<<<gAV, 256, 0, stream>>>(attn, NNODES, NN, TVt, NNODES, (long long)HDIM * NNODES,
        comb, EMB, HDIM, (bf16*)nullptr, 0, 0, 0, (const float*)nullptr,
        NNODES, 1.0f, 1 /*accumulate*/);

    // ---- output projection ----
    cvt_f32_bf16<<<cvtX, 256, 0, stream>>>(comb, combb, nX);
    wmma_gemm<<<gLin, 256, 0, stream>>>(combb, EMB, 0, Wot, EMB, 0,
        out, EMB, 0, (bf16*)nullptr, 0, 0, 0, bo, EMB, 1.0f, 0);
}
